// LightGlue_36197984371453
// MI455X (gfx1250) — matrix operand order repaired and checked by reading, not compiled, not run on hardware
//
#include <hip/hip_runtime.h>
#include <math.h>
#include <stdint.h>

// ---------------- sizes ----------------
#define NT   1024      // tokens
#define DD   256       // model dim
#define NH   4         // heads
#define DH   64        // head dim
#define FH   32        // DH/2
#define NL   9         // layers
#define D2   512       // 2*D

typedef __attribute__((ext_vector_type(16))) _Float16 v16h;
typedef __attribute__((ext_vector_type(8)))  _Float16 v8h;
typedef __attribute__((ext_vector_type(4)))  _Float16 v4h;
typedef __attribute__((ext_vector_type(8)))  float    v8f;

union AFrag { v16h v; v8h h[2]; };

#define GBM 128
#define GBN 64
#define GBK 32
#define GTH 256

// flags: 1 = add bias[n]; 2 = add resid(m,n); 4 = also write f16 shadow C16(m,n)

// ================= f32-A WMMA GEMM (used only where A is strided) =================
// C(m,n) = alpha * sum_k A(m,k)*B(k,n);  A(m,k)=A[m*sAm+k*sAk], B(k,n)=B[k*sBk+n*sBn]
template <bool AC, bool BC>
__global__ __launch_bounds__(GTH) void gemm_wmma_f32(
    const float* __restrict__ A, const float* __restrict__ B,
    float* __restrict__ C, _Float16* __restrict__ C16,
    const float* __restrict__ bias, const float* __restrict__ resid,
    int M, int N, int K,
    long sAm, long sAk, long sBk, long sBn, long sCm,
    long batA, long batB, long batC,
    float alpha, int flags)
{
    __shared__ alignas(32) _Float16 As[2][GBM * GBK];   // m-major: As[m*GBK+k]
    __shared__ alignas(32) _Float16 Bs[2][GBN * GBK];   // n-major: Bs[n*GBK+k]

    const long bz = blockIdx.z;
    A += bz * batA;  B += bz * batB;  C += bz * batC;
    if (C16) C16 += bz * batC;
    const float* R = resid ? (resid + bz * batC) : nullptr;

    const int m0   = blockIdx.y * GBM;
    const int n0   = blockIdx.x * GBN;
    const int tid  = threadIdx.x;
    const int lane = tid & 31;
    const int wave = tid >> 5;
    const int wM   = wave & 3;
    const int wN   = wave >> 2;

    const float* pA = AC ? (A + (long)(m0 + (tid >> 3)) * sAm + (long)(tid & 7) * 4)
                         : (A + (long)(m0 + (tid >> 5)) * sAm + (long)(tid & 31) * sAk);
    const float* pB = BC ? (B + (long)(tid >> 4) * sBk + (long)(n0 + (tid & 15) * 4))
                         : (B + (long)(tid >> 6) * sBk + (long)(n0 + (tid & 63)) * sBn);

    float4 a4[4]; float a1[16];
    float4 b4[2]; float b1[8];

    auto loadTiles = [&](int k0) {
        if (AC) {
#pragma unroll
            for (int r = 0; r < 4; ++r)
                a4[r] = *(const float4*)(pA + (long)k0 + (long)r * 32 * sAm);
        } else {
#pragma unroll
            for (int r = 0; r < 16; ++r)
                a1[r] = pA[(long)k0 * sAk + (long)r * 8 * sAm];
        }
        if (BC) {
#pragma unroll
            for (int r = 0; r < 2; ++r)
                b4[r] = *(const float4*)(pB + (long)(k0 + r * 16) * sBk);
        } else {
#pragma unroll
            for (int r = 0; r < 8; ++r)
                b1[r] = pB[(long)(k0 + r * 4) * sBk];
        }
    };

    auto storeTiles = [&](int buf) {
        if (AC) {
#pragma unroll
            for (int r = 0; r < 4; ++r) {
                v4h h;
                h[0] = (_Float16)a4[r].x; h[1] = (_Float16)a4[r].y;
                h[2] = (_Float16)a4[r].z; h[3] = (_Float16)a4[r].w;
                *(v4h*)(&As[buf][((tid >> 3) + r * 32) * GBK + (tid & 7) * 4]) = h;
            }
        } else {
#pragma unroll
            for (int r = 0; r < 16; ++r)
                As[buf][((tid >> 5) + r * 8) * GBK + (tid & 31)] = (_Float16)a1[r];
        }
        if (BC) {
#pragma unroll
            for (int r = 0; r < 2; ++r) {
                int kk = (tid >> 4) + r * 16;
                int nn = (tid & 15) * 4;
                Bs[buf][(nn + 0) * GBK + kk] = (_Float16)b4[r].x;
                Bs[buf][(nn + 1) * GBK + kk] = (_Float16)b4[r].y;
                Bs[buf][(nn + 2) * GBK + kk] = (_Float16)b4[r].z;
                Bs[buf][(nn + 3) * GBK + kk] = (_Float16)b4[r].w;
            }
        } else {
#pragma unroll
            for (int r = 0; r < 8; ++r)
                Bs[buf][(tid & 63) * GBK + ((tid >> 6) + r * 4)] = (_Float16)b1[r];
        }
    };

    v8f acc[2][2];
#pragma unroll
    for (int i = 0; i < 2; ++i)
#pragma unroll
        for (int j = 0; j < 2; ++j) acc[i][j] = (v8f)0.0f;

    const int arow = lane & 15;
    const int kbA  = (lane >> 4) * 16;
    const int kbB  = (lane >> 4) * 16;
    const int nsteps = K / GBK;

    loadTiles(0);
    storeTiles(0);
    __syncthreads();

    for (int s = 0; s < nsteps; ++s) {
        const int  cur     = s & 1;
        const bool hasNext = (s + 1 < nsteps);
        if (hasNext) loadTiles((s + 1) * GBK);

        AFrag af[2];
        v16h  bf[2];
#pragma unroll
        for (int mf = 0; mf < 2; ++mf) {
            int row = wM * 32 + mf * 16 + arow;
            af[mf].h[0] = *(const v8h*)(&As[cur][row * GBK + kbA]);
            af[mf].h[1] = *(const v8h*)(&As[cur][row * GBK + kbA + 8]);
        }
#pragma unroll
        for (int nf = 0; nf < 2; ++nf) {
            int col = wN * 32 + nf * 16 + (lane & 15);
            bf[nf] = *(const v16h*)(&Bs[cur][col * GBK + kbB]);
        }
#pragma unroll
        for (int mf = 0; mf < 2; ++mf)
#pragma unroll
            for (int nf = 0; nf < 2; ++nf)
                acc[mf][nf] = __builtin_amdgcn_wmma_f32_16x16x32_f16(
                    false, af[mf].v, false, bf[nf],
                    (short)0, acc[mf][nf], false, false);

        if (hasNext) {
            storeTiles(cur ^ 1);
            __syncthreads();
        }
    }

#pragma unroll
    for (int mf = 0; mf < 2; ++mf) {
#pragma unroll
        for (int nf = 0; nf < 2; ++nf) {
            int col   = n0 + wN * 32 + nf * 16 + (lane & 15);
            int rbase = m0 + wM * 32 + mf * 16 + (lane >> 4) * 8;
            float bv  = (flags & 1) ? bias[col] : 0.0f;
#pragma unroll
            for (int j = 0; j < 8; ++j) {
                long off = (long)(rbase + j) * sCm + col;
                float v = alpha * acc[mf][nf][j] + bv;
                if (flags & 2) v += R[off];
                C[off] = v;
                if (flags & 4) C16[off] = (_Float16)v;
            }
        }
    }
}

// ================= f16-A WMMA GEMM with async DMA staging of A =================
// A16 is f16 row-major (k contiguous, row stride sAm halves). The A tile is DMA'd
// straight into LDS with GLOBAL_LOAD_ASYNC_TO_LDS_B128 (ASYNCcnt) — global layout
// equals the m-major LDS layout the ISA A-fragment wants, so no VALU is spent.
// B is f32, converted in registers (co-executes with WMMA).
template <bool BC>
__global__ __launch_bounds__(GTH) void gemm_wmma_a16(
    const _Float16* __restrict__ A16, const float* __restrict__ B,
    float* __restrict__ C, _Float16* __restrict__ C16,
    const float* __restrict__ bias, const float* __restrict__ resid,
    int M, int N, int K,
    long sAm, long sBk, long sBn, long sCm,
    long batA, long batB, long batC,
    float alpha, int flags)
{
    __shared__ alignas(32) _Float16 As[2][GBM * GBK];   // m-major (matches global rows)
    __shared__ alignas(32) _Float16 Bs[2][GBN * GBK];   // n-major

    const long bz = blockIdx.z;
    A16 += bz * batA;  B += bz * batB;  C += bz * batC;
    if (C16) C16 += bz * batC;
    const float* R = resid ? (resid + bz * batC) : nullptr;

    const int m0   = blockIdx.y * GBM;
    const int n0   = blockIdx.x * GBN;
    const int tid  = threadIdx.x;
    const int lane = tid & 31;
    const int wave = tid >> 5;
    const int wM   = wave & 3;
    const int wN   = wave >> 2;

    // A tile = 128 rows x 32 halves = 512 chunks of 16B; 2 chunks per thread
    const int ca = tid, cb = tid + GTH;                  // row = c>>2, q = c&3
    const _Float16* pAa = A16 + (long)(m0 + (ca >> 2)) * sAm + (ca & 3) * 8;
    const _Float16* pAb = A16 + (long)(m0 + (cb >> 2)) * sAm + (cb & 3) * 8;
    const uint32_t ldsAa = (uint32_t)(uintptr_t)&As[0][(ca >> 2) * GBK + (ca & 3) * 8];
    const uint32_t ldsAb = (uint32_t)(uintptr_t)&As[0][(cb >> 2) * GBK + (cb & 3) * 8];
    const uint32_t bufStride = GBM * GBK * (uint32_t)sizeof(_Float16);

    const float* pB = BC ? (B + (long)(tid >> 4) * sBk + (long)(n0 + (tid & 15) * 4))
                         : (B + (long)(tid >> 6) * sBk + (long)(n0 + (tid & 63)) * sBn);
    float4 b4[2]; float b1[8];

    auto asyncA = [&](int k0, int buf) {
        uint64_t ga = (uint64_t)(uintptr_t)(pAa + k0);
        uint64_t gb = (uint64_t)(uintptr_t)(pAb + k0);
        uint32_t da = ldsAa + (uint32_t)buf * bufStride;
        uint32_t db = ldsAb + (uint32_t)buf * bufStride;
        asm volatile("global_load_async_to_lds_b128 %0, %1, off"
                     :: "v"(da), "v"(ga) : "memory");
        asm volatile("global_load_async_to_lds_b128 %0, %1, off"
                     :: "v"(db), "v"(gb) : "memory");
    };

    auto loadB = [&](int k0) {
        if (BC) {
#pragma unroll
            for (int r = 0; r < 2; ++r)
                b4[r] = *(const float4*)(pB + (long)(k0 + r * 16) * sBk);
        } else {
#pragma unroll
            for (int r = 0; r < 8; ++r)
                b1[r] = pB[(long)(k0 + r * 4) * sBk];
        }
    };

    auto storeB = [&](int buf) {
        if (BC) {
#pragma unroll
            for (int r = 0; r < 2; ++r) {
                int kk = (tid >> 4) + r * 16;
                int nn = (tid & 15) * 4;
                Bs[buf][(nn + 0) * GBK + kk] = (_Float16)b4[r].x;
                Bs[buf][(nn + 1) * GBK + kk] = (_Float16)b4[r].y;
                Bs[buf][(nn + 2) * GBK + kk] = (_Float16)b4[r].z;
                Bs[buf][(nn + 3) * GBK + kk] = (_Float16)b4[r].w;
            }
        } else {
#pragma unroll
            for (int r = 0; r < 8; ++r)
                Bs[buf][(tid & 63) * GBK + ((tid >> 6) + r * 4)] = (_Float16)b1[r];
        }
    };

    v8f acc[2][2];
#pragma unroll
    for (int i = 0; i < 2; ++i)
#pragma unroll
        for (int j = 0; j < 2; ++j) acc[i][j] = (v8f)0.0f;

    const int arow = lane & 15;
    const int kbA  = (lane >> 4) * 16;
    const int kbB  = (lane >> 4) * 16;
    const int nsteps = K / GBK;

    asyncA(0, 0);
    loadB(0);
    storeB(0);
    asm volatile("s_wait_asynccnt 0x0" ::: "memory");
    __syncthreads();

    for (int s = 0; s < nsteps; ++s) {
        const int  cur     = s & 1;
        const bool hasNext = (s + 1 < nsteps);
        if (hasNext) {
            asyncA((s + 1) * GBK, cur ^ 1);   // DMA next A tile while WMMAs run
            loadB((s + 1) * GBK);
        }

        AFrag af[2];
        v16h  bf[2];
#pragma unroll
        for (int mf = 0; mf < 2; ++mf) {
            int row = wM * 32 + mf * 16 + arow;
            af[mf].h[0] = *(const v8h*)(&As[cur][row * GBK + kbA]);
            af[mf].h[1] = *(const v8h*)(&As[cur][row * GBK + kbA + 8]);
        }
#pragma unroll
        for (int nf = 0; nf < 2; ++nf) {
            int col = wN * 32 + nf * 16 + (lane & 15);
            bf[nf] = *(const v16h*)(&Bs[cur][col * GBK + kbB]);
        }
#pragma unroll
        for (int mf = 0; mf < 2; ++mf)
#pragma unroll
            for (int nf = 0; nf < 2; ++nf)
                acc[mf][nf] = __builtin_amdgcn_wmma_f32_16x16x32_f16(
                    false, af[mf].v, false, bf[nf],
                    (short)0, acc[mf][nf], false, false);

        if (hasNext) {
            storeB(cur ^ 1);
            asm volatile("s_wait_asynccnt 0x0" ::: "memory");
            __syncthreads();
        }
    }

#pragma unroll
    for (int mf = 0; mf < 2; ++mf) {
#pragma unroll
        for (int nf = 0; nf < 2; ++nf) {
            int col   = n0 + wN * 32 + nf * 16 + (lane & 15);
            int rbase = m0 + wM * 32 + mf * 16 + (lane >> 4) * 8;
            float bv  = (flags & 1) ? bias[col] : 0.0f;
#pragma unroll
            for (int j = 0; j < 8; ++j) {
                long off = (long)(rbase + j) * sCm + col;
                float v = alpha * acc[mf][nf][j] + bv;
                if (flags & 2) v += R[off];
                C[off] = v;
                if (flags & 4) C16[off] = (_Float16)v;
            }
        }
    }
}

// ---------------- elementwise / reduction kernels ----------------

__global__ void f32_to_f16_kernel(const float* __restrict__ src, _Float16* __restrict__ dst, int n)
{
    int t = blockIdx.x * blockDim.x + threadIdx.x;
    if (t < n) dst[t] = (_Float16)src[t];
}

__global__ void posenc_kernel(const float* __restrict__ kpts, const float* __restrict__ Wr,
                              float* __restrict__ cosb, float* __restrict__ sinb)
{
    int i = blockIdx.x, k = threadIdx.x;                 // <<<NT, FH>>>
    float p = kpts[i * 2 + 0] * Wr[k] + kpts[i * 2 + 1] * Wr[FH + k];
    cosb[i * FH + k] = cosf(p);
    sinb[i * FH + k] = sinf(p);
}

// split packed qkv into merged Q/K/V (NT x 256) with interleaved RoPE on Q,K.
// Q also written as f16 shadow (A operand of the scores GEMM).
__global__ void rope_split_kernel(const float* __restrict__ qkv,
                                  const float* __restrict__ cosb, const float* __restrict__ sinb,
                                  float* __restrict__ Q, float* __restrict__ K, float* __restrict__ V,
                                  _Float16* __restrict__ Q16)
{
    int t = blockIdx.x * blockDim.x + threadIdx.x;       // NT*NH*FH threads
    if (t >= NT * NH * FH) return;
    int k = t % FH;
    int h = (t / FH) % NH;
    int n = t / (FH * NH);
    int base = n * (3 * DD) + h * DH * 3;
    int c0 = base + (2 * k) * 3;
    int c1 = base + (2 * k + 1) * 3;
    float cs = cosb[n * FH + k], sn = sinb[n * FH + k];
    float q0 = qkv[c0 + 0], q1 = qkv[c1 + 0];
    float k0 = qkv[c0 + 1], k1 = qkv[c1 + 1];
    float v0 = qkv[c0 + 2], v1 = qkv[c1 + 2];
    int o = n * DD + h * DH + 2 * k;
    float qa = q0 * cs - q1 * sn, qb = q1 * cs + q0 * sn;
    Q[o] = qa;  Q[o + 1] = qb;
    Q16[o] = (_Float16)qa;  Q16[o + 1] = (_Float16)qb;
    K[o]     = k0 * cs - k1 * sn;
    K[o + 1] = k1 * cs + k0 * sn;
    V[o]     = v0;
    V[o + 1] = v1;
}

__global__ void softmax_row_kernel(const float* __restrict__ src, float* __restrict__ dst,
                                   _Float16* __restrict__ dst16, int W)
{
    long row = blockIdx.x;
    src += row * (long)W;  dst += row * (long)W;  dst16 += row * (long)W;
    __shared__ float red[256];
    int tid = threadIdx.x;
    float mx = -1e30f;
    for (int j = tid; j < W; j += 256) mx = fmaxf(mx, src[j]);
    red[tid] = mx; __syncthreads();
    for (int s = 128; s > 0; s >>= 1) { if (tid < s) red[tid] = fmaxf(red[tid], red[tid + s]); __syncthreads(); }
    mx = red[0]; __syncthreads();
    float sum = 0.0f;
    for (int j = tid; j < W; j += 256) { float e = expf(src[j] - mx); dst[j] = e; sum += e; }
    red[tid] = sum; __syncthreads();
    for (int s = 128; s > 0; s >>= 1) { if (tid < s) red[tid] += red[tid + s]; __syncthreads(); }
    float inv = 1.0f / red[0];
    for (int j = tid; j < W; j += 256) {
        float v = dst[j] * inv;
        dst[j] = v;
        dst16[j] = (_Float16)v;
    }
}

__global__ void row_stats_kernel(const float* __restrict__ src, float* __restrict__ rmax,
                                 float* __restrict__ rsum, int W)
{
    long row = blockIdx.x;
    src += row * (long)W;
    __shared__ float red[256];
    int tid = threadIdx.x;
    float mx = -1e30f;
    for (int j = tid; j < W; j += 256) mx = fmaxf(mx, src[j]);
    red[tid] = mx; __syncthreads();
    for (int s = 128; s > 0; s >>= 1) { if (tid < s) red[tid] = fmaxf(red[tid], red[tid + s]); __syncthreads(); }
    mx = red[0]; __syncthreads();
    float sum = 0.0f;
    for (int j = tid; j < W; j += 256) sum += expf(src[j] - mx);
    red[tid] = sum; __syncthreads();
    for (int s = 128; s > 0; s >>= 1) { if (tid < s) red[tid] += red[tid + s]; __syncthreads(); }
    if (tid == 0) { rmax[row] = mx; rsum[row] = red[0]; }
}

__global__ void col_stats_kernel(const float* __restrict__ src, float* __restrict__ cmax,
                                 float* __restrict__ csum, int Rows, int W)
{
    int j = blockIdx.x * blockDim.x + threadIdx.x;
    if (j >= W) return;
    src  += (long)blockIdx.y * Rows * W;
    cmax += (long)blockIdx.y * W;
    csum += (long)blockIdx.y * W;
    float mx = -1e30f, sum = 0.0f;
    for (int i = 0; i < Rows; ++i) {
        float v = src[(long)i * W + j];
        if (v > mx) { sum = sum * expf(mx - v) + 1.0f; mx = v; }
        else        { sum += expf(v - mx); }
    }
    cmax[j] = mx; csum[j] = sum;
}

__global__ void col_softmax_apply_kernel(const float* __restrict__ src, float* __restrict__ dst,
                                         const float* __restrict__ cmax, const float* __restrict__ csum,
                                         int Rows, int W, int batch)
{
    long t = (long)blockIdx.x * blockDim.x + threadIdx.x;
    long total = (long)batch * Rows * W;
    if (t >= total) return;
    long per = (long)Rows * W;
    int  b = (int)(t / per);
    int  j = (int)(t % W);
    dst[t] = expf(src[t] - cmax[b * W + j]) / csum[b * W + j];
}

__global__ void concat_kernel(const float* __restrict__ x, const float* __restrict__ m,
                              float* __restrict__ cat, _Float16* __restrict__ cat16)
{
    int t = blockIdx.x * blockDim.x + threadIdx.x;       // NT*D2
    if (t >= NT * D2) return;
    int n = t / D2, j = t % D2;
    float v = (j < DD) ? x[n * DD + j] : m[n * DD + (j - DD)];
    cat[t] = v;
    cat16[t] = (_Float16)v;
}

__global__ void ln_gelu_kernel(float* __restrict__ x, const float* __restrict__ g,
                               const float* __restrict__ beta, int W,
                               _Float16* __restrict__ out16)
{
    long row = blockIdx.x;
    x += row * (long)W;  out16 += row * (long)W;
    __shared__ float red[256];
    __shared__ float s_mu, s_rstd;
    int tid = threadIdx.x;
    float sum = 0.0f;
    for (int j = tid; j < W; j += 256) sum += x[j];
    red[tid] = sum; __syncthreads();
    for (int s = 128; s > 0; s >>= 1) { if (tid < s) red[tid] += red[tid + s]; __syncthreads(); }
    if (tid == 0) s_mu = red[0] / W;
    __syncthreads();
    float mu = s_mu, vs = 0.0f;
    for (int j = tid; j < W; j += 256) { float d = x[j] - mu; vs += d * d; }
    red[tid] = vs; __syncthreads();
    for (int s = 128; s > 0; s >>= 1) { if (tid < s) red[tid] += red[tid + s]; __syncthreads(); }
    if (tid == 0) s_rstd = rsqrtf(red[0] / W + 1e-5f);
    __syncthreads();
    float rstd = s_rstd;
    for (int j = tid; j < W; j += 256) {
        float y = (x[j] - mu) * rstd * g[j] + beta[j];
        float r = 0.5f * y * (1.0f + erff(y * 0.70710678118654752f));   // exact GELU
        x[j] = r;
        out16[j] = (_Float16)r;
    }
}

__global__ void zproj_kernel(const float* __restrict__ d, const float* __restrict__ Wz,
                             const float* __restrict__ bz, float* __restrict__ z)
{
    long row = blockIdx.x;
    __shared__ float red[256];
    int tid = threadIdx.x;
    red[tid] = d[row * DD + tid] * Wz[tid];
    __syncthreads();
    for (int s = 128; s > 0; s >>= 1) { if (tid < s) red[tid] += red[tid + s]; __syncthreads(); }
    if (tid == 0) z[row] = red[0] + bz[0];
}

__device__ __forceinline__ float logsig(float x) {
    return fminf(x, 0.0f) - log1pf(expf(-fabsf(x)));
}

__global__ void assign_combine_kernel(const float* __restrict__ sim,
                                      const float* __restrict__ rmax, const float* __restrict__ rsum,
                                      const float* __restrict__ cmax, const float* __restrict__ csum,
                                      const float* __restrict__ z0, const float* __restrict__ z1,
                                      float* __restrict__ out)
{
    long t = (long)blockIdx.x * blockDim.x + threadIdx.x;
    const int W = NT + 1;
    if (t >= (long)W * W) return;
    int i = (int)(t / W), j = (int)(t % W);
    if (i < NT && j < NT) {
        float s  = sim[(long)i * NT + j];
        float rl = rmax[i] + logf(rsum[i]);
        float cl = cmax[j] + logf(csum[j]);
        out[t] = (s - rl) + (s - cl) + logsig(z0[i]) + logsig(z1[j]);
    } else if (i < NT) {
        out[t] = logsig(-z0[i]);
    } else if (j < NT) {
        out[t] = logsig(-z1[j]);
    } else {
        out[t] = 0.0f;
    }
}

// ---------------- host orchestration ----------------

static void gemmF32(hipStream_t s, const float* A, const float* B, float* C, _Float16* C16,
                    const float* bias, const float* resid,
                    int M, int N, int K,
                    long sAm, long sAk, long sBk, long sBn, long sCm,
                    long bA, long bB, long bC, int batch, float alpha, int flags)
{
    dim3 grid(N / GBN, M / GBM, batch);
    const bool ac = (sAk == 1), bc = (sBn == 1);
    if (ac && bc)
        gemm_wmma_f32<true, true><<<grid, dim3(GTH), 0, s>>>(A, B, C, C16, bias, resid, M, N, K,
            sAm, sAk, sBk, sBn, sCm, bA, bB, bC, alpha, flags);
    else if (ac)
        gemm_wmma_f32<true, false><<<grid, dim3(GTH), 0, s>>>(A, B, C, C16, bias, resid, M, N, K,
            sAm, sAk, sBk, sBn, sCm, bA, bB, bC, alpha, flags);
    else if (bc)
        gemm_wmma_f32<false, true><<<grid, dim3(GTH), 0, s>>>(A, B, C, C16, bias, resid, M, N, K,
            sAm, sAk, sBk, sBn, sCm, bA, bB, bC, alpha, flags);
    else
        gemm_wmma_f32<false, false><<<grid, dim3(GTH), 0, s>>>(A, B, C, C16, bias, resid, M, N, K,
            sAm, sAk, sBk, sBn, sCm, bA, bB, bC, alpha, flags);
}

static void gemmA16(hipStream_t s, const _Float16* A16, const float* B, float* C, _Float16* C16,
                    const float* bias, const float* resid,
                    int M, int N, int K,
                    long sAm, long sBk, long sBn, long sCm,
                    long bA, long bB, long bC, int batch, float alpha, int flags)
{
    dim3 grid(N / GBN, M / GBM, batch);
    if (sBn == 1)
        gemm_wmma_a16<true><<<grid, dim3(GTH), 0, s>>>(A16, B, C, C16, bias, resid, M, N, K,
            sAm, sBk, sBn, sCm, bA, bB, bC, alpha, flags);
    else
        gemm_wmma_a16<false><<<grid, dim3(GTH), 0, s>>>(A16, B, C, C16, bias, resid, M, N, K,
            sAm, sBk, sBn, sCm, bA, bB, bC, alpha, flags);
}

struct WS {
    float *x0, *x1, *cos0, *sin0, *cos1, *sin1;
    float *qkv, *Qb, *Kb, *Vb, *sc, *sm, *ctx, *msg, *cat, *h1;
    float *cmax, *csum, *rmax, *rsum, *z0, *z1;
    _Float16 *x0h, *x1h, *qh, *smh, *ctxh, *cath, *h1h;
};

static void ffn(hipStream_t s, float* x, _Float16* xh, const float* msg, WS& w,
                const float* W1, const float* b1, const float* g, const float* beta,
                const float* W2, const float* b2)
{
    concat_kernel<<<(NT * D2 + 255) / 256, 256, 0, s>>>(x, msg, w.cat, w.cath);
    gemmA16(s, w.cath, W1, w.h1, nullptr, b1, nullptr, NT, D2, D2,
            D2, D2, 1, D2, 0, 0, 0, 1, 1.0f, 1);
    ln_gelu_kernel<<<NT, 256, 0, s>>>(w.h1, g, beta, D2, w.h1h);
    // x = x + h1 @ W2 + b2   (also refresh the f16 shadow of x)
    gemmA16(s, w.h1h, W2, x, xh, b2, x, NT, DD, D2,
            D2, DD, 1, DD, 0, 0, 0, 1, 1.0f, 1 | 2 | 4);
}

static void self_block(hipStream_t s, float* x, _Float16* xh,
                       const float* cosb, const float* sinb, WS& w,
                       const float* Wqkv, const float* bqkv, const float* Wo, const float* bo,
                       const float* W1, const float* b1, const float* g, const float* beta,
                       const float* W2, const float* b2)
{
    // qkv = x @ Wqkv + bqkv
    gemmA16(s, xh, Wqkv, w.qkv, nullptr, bqkv, nullptr, NT, 3 * DD, DD,
            DD, 3 * DD, 1, 3 * DD, 0, 0, 0, 1, 1.0f, 1);
    rope_split_kernel<<<(NT * NH * FH + 255) / 256, 256, 0, s>>>(w.qkv, cosb, sinb,
                                                                 w.Qb, w.Kb, w.Vb, w.qh);
    // scores[h] = (Qh Kh^T) * DH^-0.5   (B = K^T via strides)
    gemmA16(s, w.qh, w.Kb, w.sc, nullptr, nullptr, nullptr, NT, NT, DH,
            DD, 1, DD, NT, DH, DH, (long)NT * NT, NH, 0.125f, 0);
    softmax_row_kernel<<<NH * NT, 256, 0, s>>>(w.sc, w.sm, w.smh, NT);
    // ctx[h] = attn @ Vh  (merged into NTxD; keep f16 shadow for the proj GEMM)
    gemmA16(s, w.smh, w.Vb, w.ctx, w.ctxh, nullptr, nullptr, NT, DH, NT,
            NT, DD, 1, DD, (long)NT * NT, DH, DH, NH, 1.0f, 4);
    // msg = ctx @ Wo + bo
    gemmA16(s, w.ctxh, Wo, w.msg, nullptr, bo, nullptr, NT, DD, DD,
            DD, DD, 1, DD, 0, 0, 0, 1, 1.0f, 1);
    ffn(s, x, xh, w.msg, w, W1, b1, g, beta, W2, b2);
}

extern "C" void kernel_launch(void* const* d_in, const int* in_sizes, int n_in,
                              void* d_out, int out_size, void* d_ws, size_t ws_size,
                              hipStream_t stream)
{
    (void)in_sizes; (void)n_in; (void)out_size; (void)ws_size;
    const float* desc0 = (const float*)d_in[0];
    const float* desc1 = (const float*)d_in[1];
    const float* kpts0 = (const float*)d_in[2];
    const float* kpts1 = (const float*)d_in[3];
    const float* Wr    = (const float*)d_in[4];
    const float* sWqkv = (const float*)d_in[5];
    const float* sbqkv = (const float*)d_in[6];
    const float* sWo   = (const float*)d_in[7];
    const float* sbo   = (const float*)d_in[8];
    const float* sfW1  = (const float*)d_in[9];
    const float* sfb1  = (const float*)d_in[10];
    const float* sfg   = (const float*)d_in[11];
    const float* sfbe  = (const float*)d_in[12];
    const float* sfW2  = (const float*)d_in[13];
    const float* sfb2  = (const float*)d_in[14];
    const float* cWqk  = (const float*)d_in[15];
    const float* cbqk  = (const float*)d_in[16];
    const float* cWv   = (const float*)d_in[17];
    const float* cbv   = (const float*)d_in[18];
    const float* cWo   = (const float*)d_in[19];
    const float* cbo   = (const float*)d_in[20];
    const float* cfW1  = (const float*)d_in[21];
    const float* cfb1  = (const float*)d_in[22];
    const float* cfg   = (const float*)d_in[23];
    const float* cfbe  = (const float*)d_in[24];
    const float* cfW2  = (const float*)d_in[25];
    const float* cfb2  = (const float*)d_in[26];
    const float* mWp   = (const float*)d_in[27];
    const float* mbp   = (const float*)d_in[28];
    const float* mWz   = (const float*)d_in[29];
    const float* mbz   = (const float*)d_in[30];
    float* out = (float*)d_out;

    // ---- bump-allocate workspace (f32 region, then f16 region)
    float* p = (float*)d_ws;
    WS w;
    w.x0  = p; p += NT * DD;
    w.x1  = p; p += NT * DD;
    w.cos0 = p; p += NT * FH;  w.sin0 = p; p += NT * FH;
    w.cos1 = p; p += NT * FH;  w.sin1 = p; p += NT * FH;
    w.qkv = p; p += NT * 3 * DD;
    w.Qb  = p; p += NT * DD;
    w.Kb  = p; p += NT * DD;
    w.Vb  = p; p += NT * DD;
    w.sc  = p; p += (long)NH * NT * NT;
    w.sm  = p; p += (long)NH * NT * NT;
    w.ctx = p; p += NT * DD;
    w.msg = p; p += NT * DD;
    w.cat = p; p += NT * D2;
    w.h1  = p; p += NT * D2;
    w.cmax = p; p += NH * NT;  w.csum = p; p += NH * NT;
    w.rmax = p; p += NT;       w.rsum = p; p += NT;
    w.z0   = p; p += NT;       w.z1   = p; p += NT;
    _Float16* ph = (_Float16*)p;
    w.x0h  = ph; ph += NT * DD;
    w.x1h  = ph; ph += NT * DD;
    w.qh   = ph; ph += NT * DD;
    w.smh  = ph; ph += (long)NH * NT * NT;
    w.ctxh = ph; ph += NT * DD;
    w.cath = ph; ph += NT * D2;
    w.h1h  = ph; ph += NT * D2;

    (void)hipMemcpyAsync(w.x0, desc0, NT * DD * sizeof(float), hipMemcpyDeviceToDevice, stream);
    (void)hipMemcpyAsync(w.x1, desc1, NT * DD * sizeof(float), hipMemcpyDeviceToDevice, stream);
    f32_to_f16_kernel<<<(NT * DD + 255) / 256, 256, 0, stream>>>(desc0, w.x0h, NT * DD);
    f32_to_f16_kernel<<<(NT * DD + 255) / 256, 256, 0, stream>>>(desc1, w.x1h, NT * DD);

    posenc_kernel<<<NT, FH, 0, stream>>>(kpts0, Wr, w.cos0, w.sin0);
    posenc_kernel<<<NT, FH, 0, stream>>>(kpts1, Wr, w.cos1, w.sin1);

    for (int i = 0; i < NL; ++i) {
        const float* sW1 = sfW1 + (long)i * D2 * D2;
        const float* sB1 = sfb1 + (long)i * D2;
        const float* sG  = sfg  + (long)i * D2;
        const float* sBe = sfbe + (long)i * D2;
        const float* sW2 = sfW2 + (long)i * D2 * DD;
        const float* sB2 = sfb2 + (long)i * DD;

        self_block(stream, w.x0, w.x0h, w.cos0, w.sin0, w,
                   sWqkv + (long)i * DD * 3 * DD, sbqkv + (long)i * 3 * DD,
                   sWo + (long)i * DD * DD, sbo + (long)i * DD,
                   sW1, sB1, sG, sBe, sW2, sB2);
        self_block(stream, w.x1, w.x1h, w.cos1, w.sin1, w,
                   sWqkv + (long)i * DD * 3 * DD, sbqkv + (long)i * 3 * DD,
                   sWo + (long)i * DD * DD, sbo + (long)i * DD,
                   sW1, sB1, sG, sBe, sW2, sB2);

        // -------- cross block --------
        const float* Wqk = cWqk + (long)i * DD * DD;  const float* bqk = cbqk + (long)i * DD;
        const float* Wv  = cWv  + (long)i * DD * DD;  const float* bv  = cbv  + (long)i * DD;
        const float* Wo  = cWo  + (long)i * DD * DD;  const float* bo  = cbo  + (long)i * DD;
        const float* cW1 = cfW1 + (long)i * D2 * D2;  const float* cB1 = cfb1 + (long)i * D2;
        const float* cG  = cfg  + (long)i * D2;       const float* cBe = cfbe + (long)i * D2;
        const float* cW2 = cfW2 + (long)i * D2 * DD;  const float* cB2 = cfb2 + (long)i * DD;

        float* qk0 = w.Qb;  float* qk1 = w.Kb;  float* v0b = w.Vb;  float* v1b = w.qkv;
        // qk0 gets an f16 shadow (A of the sim GEMM); qk1 is consumed strided as f32 B
        gemmA16(stream, w.x0h, Wqk, qk0, w.qh, bqk, nullptr, NT, DD, DD,
                DD, DD, 1, DD, 0, 0, 0, 1, 1.0f, 1 | 4);
        gemmA16(stream, w.x1h, Wqk, qk1, nullptr, bqk, nullptr, NT, DD, DD,
                DD, DD, 1, DD, 0, 0, 0, 1, 1.0f, 1);
        gemmA16(stream, w.x0h, Wv, v0b, nullptr, bv, nullptr, NT, DD, DD,
                DD, DD, 1, DD, 0, 0, 0, 1, 1.0f, 1);
        gemmA16(stream, w.x1h, Wv, v1b, nullptr, bv, nullptr, NT, DD, DD,
                DD, DD, 1, DD, 0, 0, 0, 1, 1.0f, 1);
        // sim[h] = s^2 * qk0 qk1^T, alpha = DH^-0.5 = 0.125
        gemmA16(stream, w.qh, qk1, w.sc, nullptr, nullptr, nullptr, NT, NT, DH,
                DD, 1, DD, NT, DH, DH, (long)NT * NT, NH, 0.125f, 0);
        // m0 = softmax_rows(sim) @ v1
        softmax_row_kernel<<<NH * NT, 256, 0, stream>>>(w.sc, w.sm, w.smh, NT);
        gemmA16(stream, w.smh, v1b, w.ctx, w.ctxh, nullptr, nullptr, NT, DH, NT,
                NT, DD, 1, DD, (long)NT * NT, DH, DH, NH, 1.0f, 4);
        gemmA16(stream, w.ctxh, Wo, w.msg, nullptr, bo, nullptr, NT, DD, DD,
                DD, DD, 1, DD, 0, 0, 0, 1, 1.0f, 1);
        ffn(stream, w.x0, w.x0h, w.msg, w, cW1, cB1, cG, cBe, cW2, cB2);
        // m1 = softmax_cols(sim)^T @ v0   (A is strided -> f32 GEMM path)
        col_stats_kernel<<<dim3(NT / 256, NH), 256, 0, stream>>>(w.sc, w.cmax, w.csum, NT, NT);
        col_softmax_apply_kernel<<<(int)(((long)NH * NT * NT + 255) / 256), 256, 0, stream>>>(
            w.sc, w.sm, w.cmax, w.csum, NT, NT, NH);
        gemmF32(stream, w.sm, v0b, w.ctx, w.ctxh, nullptr, nullptr, NT, DH, NT,
                1, NT, DD, 1, DD, (long)NT * NT, DH, DH, NH, 1.0f, 4);
        gemmA16(stream, w.ctxh, Wo, w.msg, nullptr, bo, nullptr, NT, DD, DD,
                DD, DD, 1, DD, 0, 0, 0, 1, 1.0f, 1);
        ffn(stream, w.x1, w.x1h, w.msg, w, cW1, cB1, cG, cBe, cW2, cB2);
    }

    // -------- assignment head --------
    float* md0 = w.Qb; float* md1 = w.Kb;
    gemmA16(stream, w.x0h, mWp, md0, w.qh, mbp, nullptr, NT, DD, DD,
            DD, DD, 1, DD, 0, 0, 0, 1, 1.0f, 1 | 4);
    gemmA16(stream, w.x1h, mWp, md1, nullptr, mbp, nullptr, NT, DD, DD,
            DD, DD, 1, DD, 0, 0, 0, 1, 1.0f, 1);
    // sim = (md0*s)(md1*s)^T, alpha = D^-0.5 = 0.0625
    gemmA16(stream, w.qh, md1, w.sc, nullptr, nullptr, nullptr, NT, NT, DD,
            DD, 1, DD, NT, 0, 0, 0, 1, 0.0625f, 0);
    zproj_kernel<<<NT, 256, 0, stream>>>(w.x0, mWz, mbz, w.z0);
    zproj_kernel<<<NT, 256, 0, stream>>>(w.x1, mWz, mbz, w.z1);
    row_stats_kernel<<<NT, 256, 0, stream>>>(w.sc, w.rmax, w.rsum, NT);
    col_stats_kernel<<<dim3(NT / 256, 1), 256, 0, stream>>>(w.sc, w.cmax, w.csum, NT, NT);
    assign_combine_kernel<<<(int)(((long)(NT + 1) * (NT + 1) + 255) / 256), 256, 0, stream>>>(
        w.sc, w.rmax, w.rsum, w.cmax, w.csum, w.z0, w.z1, out);
}
